// MambaBottleneckDeep_58342835749242
// MI455X (gfx1250) — matrix operand order repaired and checked by reading, not compiled, not run on hardware
//
#include <hip/hip_runtime.h>

#define B_   8
#define C_   256
#define HW_  1024
#define L_   1024
#define M_   8192          // B_*L_
#define DI_  512
#define DS_  16
#define DTR_ 16
#define NL_  4
#define KC_  4

typedef __attribute__((ext_vector_type(16))) __bf16 v16bf;
typedef __attribute__((ext_vector_type(8)))  float  v8f;
typedef __attribute__((__vector_size__(16))) int    v4i_;

static __device__ __forceinline__ unsigned short f2bf(float f) {
  unsigned u = __float_as_uint(f);
  u += 0x7FFFu + ((u >> 16) & 1u);      // round-to-nearest-even
  return (unsigned short)(u >> 16);
}
static __device__ __forceinline__ float sigmoidf_(float x) { return 1.0f / (1.0f + __expf(-x)); }
static __device__ __forceinline__ float siluf_(float x)    { return x * sigmoidf_(x); }

// ---------------------------------------------------------------------------
// Async global->LDS staging (CDNA5 GLOBAL_LOAD_ASYNC_TO_LDS_B128, ASYNCcnt),
// with a synchronous generic-store fallback if the builtin is unavailable.
// Probe-derived prototype: (v4i as1* src, v4i as3* dst, imm offset, imm cpol)
// ---------------------------------------------------------------------------
static __device__ __forceinline__ void tile_ld_b128(const unsigned short* gsrc,
                                                    unsigned short* ldst) {
#if defined(__gfx1250__) && __has_builtin(__builtin_amdgcn_global_load_async_to_lds_b128)
  __builtin_amdgcn_global_load_async_to_lds_b128(
      (__attribute__((address_space(1))) v4i_*)gsrc,
      (__attribute__((address_space(3))) v4i_*)ldst, 0, 0);
#else
  *reinterpret_cast<uint4*>(ldst) = *reinterpret_cast<const uint4*>(gsrc);
#endif
}
static __device__ __forceinline__ void tile_ld_wait() {
#if defined(__gfx1250__) && __has_builtin(__builtin_amdgcn_s_wait_asynccnt)
  __builtin_amdgcn_s_wait_asynccnt(0);
#endif
}

// ---------------------------------------------------------------------------
// bf16 WMMA GEMM:  out[m,n] = sum_k A[m,k] * W[n,k]   (f32 accumulate)
// A: M x K bf16 row-major, W: N x K bf16 row-major. K%32==0, N%64==0, M%128==0
// Block = 256 threads (8 waves). M-tile 128 (16 rows/wave), N-tile 64.
// W tile (64x32 bf16) double-buffered in LDS via async load, 4 WMMA / K-step.
// ---------------------------------------------------------------------------
__global__ __launch_bounds__(256) void gemm_bf16_wmma(
    const unsigned short* __restrict__ A,
    const unsigned short* __restrict__ W,
    float* __restrict__ out,
    int M, int N, int K)
{
  __shared__ unsigned short lB[2][64 * 32];  // [buf][n_local*32 + k_local]

  const int wave = threadIdx.x >> 5;
  const int lane = threadIdx.x & 31;
  const int hl   = lane >> 4;                // half of wave (0/1)
  const int l16  = lane & 15;

  const int mBase = blockIdx.y * 128 + wave * 16;
  const int nBase = blockIdx.x * 64;
  const int aRow  = mBase + l16;             // A-fragment row for this lane

  // cooperative W-tile load mapping: 256 threads x 16B = 64 rows x 64B
  const int ldN = threadIdx.x >> 2;          // 0..63
  const int ldK = (threadIdx.x & 3) * 8;     // 0,8,16,24
  const unsigned short* wRow = W + (size_t)(nBase + ldN) * K + ldK;
  unsigned short* ld0 = &lB[0][ldN * 32 + ldK];
  unsigned short* ld1 = &lB[1][ldN * 32 + ldK];

  v8f acc0 = {}, acc1 = {}, acc2 = {}, acc3 = {};

  union AF { v16bf v; uint4 q[2]; };
  union BF { v16bf v; uint4 q[2]; };

  const int nk = K / 32;

  // prologue: stage tile 0 into buffer 0
  tile_ld_b128(wRow, ld0);
  tile_ld_wait();
  __syncthreads();

  for (int ki = 0; ki < nk; ++ki) {
    const int cur = ki & 1;
    if (ki + 1 < nk)                         // prefetch next W tile into other buf
      tile_ld_b128(wRow + (ki + 1) * 32, cur ? ld0 : ld1);

    // A fragment: lane<16 holds K = {0..7, 16..23}, lane>=16 holds {8..15, 24..31}
    AF af;
    const unsigned short* ap = A + (size_t)aRow * K + ki * 32 + hl * 8;
    af.q[0] = *reinterpret_cast<const uint4*>(ap);
    af.q[1] = *reinterpret_cast<const uint4*>(ap + 16);
    if (ki + 1 < nk) __builtin_prefetch(ap + 32, 0, 0);

    // 4 B fragments from current LDS buffer (ds_load_b128 pairs)
    BF bfr[4];
#pragma unroll
    for (int s = 0; s < 4; ++s) {
      const unsigned short* bp = &lB[cur][(s * 16 + l16) * 32 + hl * 8];
      bfr[s].q[0] = *reinterpret_cast<const uint4*>(bp);
      bfr[s].q[1] = *reinterpret_cast<const uint4*>(bp + 16);
    }

    acc0 = __builtin_amdgcn_wmma_f32_16x16x32_bf16(false, af.v, false, bfr[0].v,
                                                   (short)0, acc0, false, false);
    acc1 = __builtin_amdgcn_wmma_f32_16x16x32_bf16(false, af.v, false, bfr[1].v,
                                                   (short)0, acc1, false, false);
    acc2 = __builtin_amdgcn_wmma_f32_16x16x32_bf16(false, af.v, false, bfr[2].v,
                                                   (short)0, acc2, false, false);
    acc3 = __builtin_amdgcn_wmma_f32_16x16x32_bf16(false, af.v, false, bfr[3].v,
                                                   (short)0, acc3, false, false);

    tile_ld_wait();                          // async writes to next buf complete
    __syncthreads();                         // all reads of cur buf complete
  }

  // C/D layout: lane<16 -> M = e, N = lane ; lane>=16 -> M = 8+e, N = lane-16
  const int mRow0 = mBase + hl * 8;
#pragma unroll
  for (int e = 0; e < 8; ++e) {
    size_t r = (size_t)(mRow0 + e) * N + nBase + l16;
    out[r]      = acc0[e];
    out[r + 16] = acc1[e];
    out[r + 32] = acc2[e];
    out[r + 48] = acc3[e];
  }
}

// ---------------------------------------------------------------------------
// fp32 weights (N x K) -> padded bf16 (Npad x Kpad), zero-filled
// ---------------------------------------------------------------------------
__global__ void cvt_w_bf16(const float* __restrict__ src, unsigned short* __restrict__ dst,
                           int N, int K, int Npad, int Kpad)
{
  int idx = blockIdx.x * blockDim.x + threadIdx.x;
  if (idx >= Npad * Kpad) return;
  int n = idx / Kpad, k = idx % Kpad;
  float v = (n < N && k < K) ? src[(size_t)n * K + k] : 0.0f;
  dst[idx] = f2bf(v);
}

// xs[b, l, c] = x[b, c, l]
__global__ void init_xs(const float* __restrict__ x, float* __restrict__ xs)
{
  int idx = blockIdx.x * blockDim.x + threadIdx.x;
  if (idx >= B_ * C_ * HW_) return;
  int b = idx / (C_ * HW_);
  int c = (idx / HW_) % C_;
  int l = idx % HW_;
  xs[((size_t)b * L_ + l) * C_ + c] = x[idx];
}

// LayerNorm over C per token -> bf16 A matrix (M x C)
__global__ __launch_bounds__(256) void layernorm_bf16(
    const float* __restrict__ xs, const float* __restrict__ lw,
    const float* __restrict__ lb, unsigned short* __restrict__ outbf)
{
  __shared__ float red[256];
  int m = blockIdx.x;
  int c = threadIdx.x;
  float v = xs[(size_t)m * C_ + c];

  red[c] = v;  __syncthreads();
  for (int s = 128; s > 0; s >>= 1) { if (c < s) red[c] += red[c + s]; __syncthreads(); }
  float mu = red[0] * (1.0f / C_);  __syncthreads();

  float d = v - mu;
  red[c] = d * d;  __syncthreads();
  for (int s = 128; s > 0; s >>= 1) { if (c < s) red[c] += red[c + s]; __syncthreads(); }
  float rstd = rsqrtf(red[0] * (1.0f / C_) + 1e-5f);

  outbf[(size_t)m * C_ + c] = f2bf(d * rstd * lw[c] + lb[c]);
}

// depthwise causal conv (K=4) + bias + SiLU; reads in_proj output cols [0,DI)
__global__ void conv_silu(const float* __restrict__ g1, const float* __restrict__ cw,
                          const float* __restrict__ cb, float* __restrict__ xin,
                          unsigned short* __restrict__ xinbf)
{
  int idx = blockIdx.x * blockDim.x + threadIdx.x;
  if (idx >= M_ * DI_) return;
  int m = idx / DI_, d = idx % DI_;
  int l = m % L_;
  float s = cb[d];
#pragma unroll
  for (int j = 0; j < KC_; ++j) {
    int ll = l - (KC_ - 1) + j;
    if (ll >= 0) s += cw[d * KC_ + j] * g1[(size_t)(m - (KC_ - 1) + j) * 1024 + d];
  }
  float v = siluf_(s);
  xin[idx]   = v;
  xinbf[idx] = f2bf(v);
}

// build dt GEMM A: M x 32 bf16 (cols 0..15 = proj[:, :16], rest zero)
__global__ void build_dtin(const float* __restrict__ proj, unsigned short* __restrict__ dtin)
{
  int idx = blockIdx.x * blockDim.x + threadIdx.x;
  if (idx >= M_ * 32) return;
  int m = idx / 32, k = idx % 32;
  dtin[idx] = f2bf(k < DTR_ ? proj[(size_t)m * 64 + k] : 0.0f);
}

// dt = softplus(dtlin + db[d])  (in place)
__global__ void softplus_bias(float* __restrict__ dt, const float* __restrict__ db)
{
  int idx = blockIdx.x * blockDim.x + threadIdx.x;
  if (idx >= M_ * DI_) return;
  float v = dt[idx] + db[idx % DI_];
  dt[idx] = (v > 20.0f) ? v : log1pf(__expf(v));
}

// selective scan: block = 64 channels of one batch; grid (B, DI/64).
// Sequential over L, state h[16] in registers; B/C chunks staged in LDS.
__global__ __launch_bounds__(64) void scan_kernel(
    const float* __restrict__ proj, const float* __restrict__ dt,
    const float* __restrict__ xin, const float* __restrict__ alog,
    const float* __restrict__ Dp, float* __restrict__ y)
{
  __shared__ float sB[32 * DS_];
  __shared__ float sC[32 * DS_];
  int b = blockIdx.x;
  int d = blockIdx.y * 64 + threadIdx.x;

  float Arow[DS_];
#pragma unroll
  for (int n = 0; n < DS_; ++n) Arow[n] = -__expf(alog[d * DS_ + n]);
  float Dd = Dp[d];
  float h[DS_];
#pragma unroll
  for (int n = 0; n < DS_; ++n) h[n] = 0.0f;

  for (int chunk = 0; chunk < L_ / 32; ++chunk) {
    for (int i = threadIdx.x; i < 32 * DS_; i += 64) {
      int s = i >> 4, n = i & 15;
      size_t row = ((size_t)b * L_ + chunk * 32 + s) * 64;
      sB[i] = proj[row + DTR_ + n];
      sC[i] = proj[row + DTR_ + DS_ + n];
    }
    __syncthreads();

    for (int st = 0; st < 32; ++st) {
      size_t m = (size_t)b * L_ + chunk * 32 + st;
      float dtv = dt[m * DI_ + d];
      float xv  = xin[m * DI_ + d];
      float dx  = dtv * xv;
      float acc = 0.0f;
#pragma unroll
      for (int nn = 0; nn < DS_; ++nn) {
        h[nn] = h[nn] * __expf(dtv * Arow[nn]) + dx * sB[st * DS_ + nn];
        acc += h[nn] * sC[st * DS_ + nn];
      }
      y[m * DI_ + d] = acc + xv * Dd;
    }
    __syncthreads();
  }
}

// ygate = bf16(y * silu(z)),  z = in_proj output cols [DI, 2*DI)
__global__ void gate_bf16(const float* __restrict__ y, const float* __restrict__ g1,
                          unsigned short* __restrict__ outbf)
{
  int idx = blockIdx.x * blockDim.x + threadIdx.x;
  if (idx >= M_ * DI_) return;
  int m = idx / DI_, d = idx % DI_;
  float z = g1[(size_t)m * 1024 + DI_ + d];
  outbf[idx] = f2bf(y[idx] * siluf_(z));
}

__global__ void residual_add(float* __restrict__ xs, const float* __restrict__ delta)
{
  int idx = blockIdx.x * blockDim.x + threadIdx.x;
  if (idx >= M_ * C_) return;
  xs[idx] += delta[idx];
}

// SE: g[b,c] = mean_l xs[b,l,c]
__global__ __launch_bounds__(256) void se_reduce(const float* __restrict__ xs, float* __restrict__ g)
{
  __shared__ float red[256];
  int bc = blockIdx.x;               // b*C_ + c
  int b = bc / C_, c = bc % C_;
  float s = 0.0f;
  for (int l = threadIdx.x; l < L_; l += 256)
    s += xs[((size_t)b * L_ + l) * C_ + c];
  red[threadIdx.x] = s;  __syncthreads();
  for (int st = 128; st > 0; st >>= 1) { if (threadIdx.x < st) red[threadIdx.x] += red[threadIdx.x + st]; __syncthreads(); }
  if (threadIdx.x == 0) g[bc] = red[0] * (1.0f / L_);
}

__global__ __launch_bounds__(256) void se_mlp(
    const float* __restrict__ g, const float* __restrict__ w1, const float* __restrict__ b1,
    const float* __restrict__ w2, const float* __restrict__ b2, float* __restrict__ g2)
{
  __shared__ float g1s[B_ * 64];
  for (int o = threadIdx.x; o < B_ * 64; o += 256) {
    int b = o / 64, j = o % 64;
    float s = b1[j];
    for (int c = 0; c < C_; ++c) s += g[b * C_ + c] * w1[j * C_ + c];
    g1s[o] = fmaxf(s, 0.0f);
  }
  __syncthreads();
  for (int o = threadIdx.x; o < B_ * C_; o += 256) {
    int b = o / C_, c = o % C_;
    float s = b2[c];
    for (int j = 0; j < 64; ++j) s += g1s[b * 64 + j] * w2[c * 64 + j];
    g2[o] = sigmoidf_(s);
  }
}

// y[b,c,l] = xs[b,l,c]*g2[b,c] + x[b,c,l]   (BCHW layout)
__global__ void se_apply(const float* __restrict__ xs, const float* __restrict__ g2,
                         const float* __restrict__ x, float* __restrict__ ybn)
{
  int idx = blockIdx.x * blockDim.x + threadIdx.x;
  if (idx >= B_ * C_ * HW_) return;
  int b = idx / (C_ * HW_);
  int c = (idx / HW_) % C_;
  int l = idx % HW_;
  ybn[idx] = xs[((size_t)b * L_ + l) * C_ + c] * g2[b * C_ + c] + x[idx];
}

// per-channel mean / inv-std over (B,H,W)
__global__ __launch_bounds__(256) void bn_stats(const float* __restrict__ ybn, float* __restrict__ stats)
{
  __shared__ float rs[256], rq[256];
  int c = blockIdx.x;
  float s = 0.0f, q = 0.0f;
  for (int i = threadIdx.x; i < B_ * HW_; i += 256) {
    int b = i / HW_, l = i % HW_;
    float v = ybn[((size_t)b * C_ + c) * HW_ + l];
    s += v; q += v * v;
  }
  rs[threadIdx.x] = s; rq[threadIdx.x] = q;  __syncthreads();
  for (int st = 128; st > 0; st >>= 1) {
    if (threadIdx.x < st) { rs[threadIdx.x] += rs[threadIdx.x + st]; rq[threadIdx.x] += rq[threadIdx.x + st]; }
    __syncthreads();
  }
  if (threadIdx.x == 0) {
    float mu  = rs[0] * (1.0f / (B_ * HW_));
    float var = rq[0] * (1.0f / (B_ * HW_)) - mu * mu;
    stats[c * 2]     = mu;
    stats[c * 2 + 1] = rsqrtf(var + 1e-5f);
  }
}

__global__ void bn_final(const float* __restrict__ ybn, const float* __restrict__ stats,
                         const float* __restrict__ bw, const float* __restrict__ bb,
                         float* __restrict__ out)
{
  int idx = blockIdx.x * blockDim.x + threadIdx.x;
  if (idx >= B_ * C_ * HW_) return;
  int c = (idx / HW_) % C_;
  out[idx] = (ybn[idx] - stats[c * 2]) * stats[c * 2 + 1] * bw[c] + bb[c];
}

// ---------------------------------------------------------------------------
extern "C" void kernel_launch(void* const* d_in, const int* in_sizes, int n_in,
                              void* d_out, int out_size, void* d_ws, size_t ws_size,
                              hipStream_t stream)
{
  const float* x      = (const float*)d_in[0];
  const float* ln_w   = (const float*)d_in[1];
  const float* ln_b   = (const float*)d_in[2];
  const float* ipw    = (const float*)d_in[3];
  const float* cw     = (const float*)d_in[4];
  const float* cb     = (const float*)d_in[5];
  const float* xpw    = (const float*)d_in[6];
  const float* dpw    = (const float*)d_in[7];
  const float* dpb    = (const float*)d_in[8];
  const float* alog   = (const float*)d_in[9];
  const float* Dp     = (const float*)d_in[10];
  const float* opw    = (const float*)d_in[11];
  const float* sew1   = (const float*)d_in[12];
  const float* seb1   = (const float*)d_in[13];
  const float* sew2   = (const float*)d_in[14];
  const float* seb2   = (const float*)d_in[15];
  const float* bnw    = (const float*)d_in[16];
  const float* bnb    = (const float*)d_in[17];
  float* out = (float*)d_out;

  char* ws = (char*)d_ws;
  size_t off = 0;
  auto alloc = [&](size_t bytes) -> char* {
    char* p = ws + off;
    off += (bytes + 255) & ~(size_t)255;
    return p;
  };
  float*          xs    = (float*)         alloc((size_t)M_ * C_  * 4);
  unsigned short* abf   = (unsigned short*)alloc((size_t)M_ * DI_ * 2);
  unsigned short* wbf   = (unsigned short*)alloc((size_t)1024 * 256 * 2);
  float*          g1out = (float*)         alloc((size_t)M_ * 1024 * 4);
  float*          xin   = (float*)         alloc((size_t)M_ * DI_ * 4);
  float*          proj  = (float*)         alloc((size_t)M_ * 64  * 4);
  float*          dtb   = (float*)         alloc((size_t)M_ * DI_ * 4);
  float*          ybuf  = (float*)         alloc((size_t)M_ * DI_ * 4);
  float*          delta = (float*)         alloc((size_t)M_ * C_  * 4);   // reused as ybn
  float*          gse   = (float*)         alloc((size_t)B_ * C_ * 4);
  float*          g2    = (float*)         alloc((size_t)B_ * C_ * 4);
  float*          stats = (float*)         alloc((size_t)C_ * 2 * 4);
  float*          ybn   = delta;

  const int TB = 256;
  const int nBCHW = B_ * C_ * HW_;

  init_xs<<<(nBCHW + TB - 1) / TB, TB, 0, stream>>>(x, xs);

  for (int i = 0; i < NL_; ++i) {
    // --- layernorm -> bf16 A ---
    layernorm_bf16<<<M_, 256, 0, stream>>>(xs, ln_w + i * C_, ln_b + i * C_, abf);

    // --- in_proj GEMM: M x 1024, K=256 ---
    cvt_w_bf16<<<(1024 * 256 + TB - 1) / TB, TB, 0, stream>>>(
        ipw + (size_t)i * 1024 * 256, wbf, 1024, 256, 1024, 256);
    gemm_bf16_wmma<<<dim3(1024 / 64, M_ / 128), 256, 0, stream>>>(abf, wbf, g1out, M_, 1024, 256);

    // --- conv + SiLU ---
    conv_silu<<<(M_ * DI_ + TB - 1) / TB, TB, 0, stream>>>(
        g1out, cw + (size_t)i * DI_ * KC_, cb + (size_t)i * DI_, xin, abf);

    // --- x_proj GEMM: M x 64 (48 padded), K=512 ---
    cvt_w_bf16<<<(64 * 512 + TB - 1) / TB, TB, 0, stream>>>(
        xpw + (size_t)i * 48 * 512, wbf, 48, 512, 64, 512);
    gemm_bf16_wmma<<<dim3(64 / 64, M_ / 128), 256, 0, stream>>>(abf, wbf, proj, M_, 64, 512);

    // --- dt_proj GEMM: M x 512, K=32 (16 padded) ---
    build_dtin<<<(M_ * 32 + TB - 1) / TB, TB, 0, stream>>>(proj, abf);
    cvt_w_bf16<<<(512 * 32 + TB - 1) / TB, TB, 0, stream>>>(
        dpw + (size_t)i * DI_ * DTR_, wbf, DI_, DTR_, DI_, 32);
    gemm_bf16_wmma<<<dim3(512 / 64, M_ / 128), 256, 0, stream>>>(abf, wbf, dtb, M_, 512, 32);
    softplus_bias<<<(M_ * DI_ + TB - 1) / TB, TB, 0, stream>>>(dtb, dpb + (size_t)i * DI_);

    // --- selective scan (64 WGPs) ---
    scan_kernel<<<dim3(B_, DI_ / 64), 64, 0, stream>>>(
        proj, dtb, xin, alog + (size_t)i * DI_ * DS_, Dp + (size_t)i * DI_, ybuf);

    // --- gate + out_proj GEMM: M x 256, K=512 ---
    gate_bf16<<<(M_ * DI_ + TB - 1) / TB, TB, 0, stream>>>(ybuf, g1out, abf);
    cvt_w_bf16<<<(256 * 512 + TB - 1) / TB, TB, 0, stream>>>(
        opw + (size_t)i * C_ * DI_, wbf, C_, DI_, C_, DI_);
    gemm_bf16_wmma<<<dim3(256 / 64, M_ / 128), 256, 0, stream>>>(abf, wbf, delta, M_, 256, 512);

    residual_add<<<(M_ * C_ + TB - 1) / TB, TB, 0, stream>>>(xs, delta);
  }

  // --- SE + residual + batchnorm ---
  se_reduce<<<B_ * C_, 256, 0, stream>>>(xs, gse);
  se_mlp<<<1, 256, 0, stream>>>(gse, sew1, seb1, sew2, seb2, g2);
  se_apply<<<(nBCHW + TB - 1) / TB, TB, 0, stream>>>(xs, g2, x, ybn);
  bn_stats<<<C_, 256, 0, stream>>>(ybn, stats);
  bn_final<<<(nBCHW + TB - 1) / TB, TB, 0, stream>>>(ybn, stats, bnw, bnb, out);

  (void)in_sizes; (void)n_in; (void)out_size; (void)ws_size;
}
